// InjectedSODA_SVD_41497974014138
// MI455X (gfx1250) — matrix operand order repaired and checked by reading, not compiled, not run on hardware
//
#include <hip/hip_runtime.h>
#include <hip/hip_bf16.h>

// ---------------------------------------------------------------------------
// CDNA5 / gfx1250 implementation.
//  out = x @ (kron(Q1,Q2,Q3) @ ((U*relu(S+delta)) @ V)^T)
// Pipeline:
//   1) Ubf  = bf16(U)                         [3072,3072]   (row-major o,k)
//   2) VsT  = bf16(relu(S+d)[k] * V[k,i])^T   [3072,3072]   (row-major i,k)
//   3) M1t[o,i] = sum_k Ubf[o,k]*VsT[i,k]     (WMMA bf16 GEMM, f32 out)
//   4) Kron stages: contract Q3 (s=1,q=12), Q2 (s=12,q=16), Q1 (s=192,q=16)
//      along the i axis of each row o; last stage emits bf16 WtT[o,i].
//   5) out[r,o] = sum_i x[r,i]*WtT[o,i]       (WMMA bf16 GEMM, x cvt on load)
// ---------------------------------------------------------------------------

typedef __bf16 bf16_t;
typedef __attribute__((ext_vector_type(16))) __bf16 v16bf;
typedef __attribute__((ext_vector_type(8)))  float  v8f;

#define DIM 3072

// ----------------------------- prep kernels --------------------------------

__global__ __launch_bounds__(256) void cvt_f32_bf16(const float* __restrict__ in,
                                                    bf16_t* __restrict__ out, int n) {
  int i = blockIdx.x * 256 + threadIdx.x;
  if (i < n) out[i] = (bf16_t)in[i];
}

// VsT[i,k] = bf16( relu(S[k]+delta[k]) * V[k,i] ), via 32x32 LDS tile transpose.
__global__ __launch_bounds__(256) void scale_transpose(const float* __restrict__ V,
                                                       const float* __restrict__ S,
                                                       const float* __restrict__ delta,
                                                       bf16_t* __restrict__ VsT) {
  __shared__ float tile[32][33];
  const int kb = blockIdx.x * 32;   // k tile base
  const int ib = blockIdx.y * 32;   // i tile base
  const int tx = threadIdx.x;       // 0..31
  const int ty = threadIdx.y;       // 0..7
#pragma unroll
  for (int r = ty; r < 32; r += 8) {
    const int k = kb + r;
    float sp = S[k] + delta[k];
    sp = sp > 0.f ? sp : 0.f;
    tile[r][tx] = V[(size_t)k * DIM + (ib + tx)] * sp;   // coalesced read
  }
  __syncthreads();
#pragma unroll
  for (int r = ty; r < 32; r += 8) {
    VsT[(size_t)(ib + r) * DIM + (kb + tx)] = (bf16_t)tile[tx][r];  // coalesced write
  }
}

// ------------------------- Kron factor contraction -------------------------
// i = hi*(q*s) + r*s + lo ;  out[o,i] = sum_j Q[r,j] * in[o, hi*q*s + j*s + lo]
template <typename OUT_T>
__global__ __launch_bounds__(256) void kron_stage(const float* __restrict__ Q,
                                                  const float* __restrict__ in,
                                                  OUT_T* __restrict__ out,
                                                  int q, int s) {
  const int i = blockIdx.x * 256 + threadIdx.x;
  if (i >= DIM) return;
  const int o  = blockIdx.y;
  const int lo = i % s;
  const int r  = (i / s) % q;
  const int hi = i / (s * q);
  const float* row = in + (size_t)o * DIM;
  const int base = hi * q * s + lo;
  float acc = 0.f;
  for (int j = 0; j < q; ++j)
    acc = fmaf(Q[r * q + j], row[base + j * s], acc);
  out[(size_t)o * DIM + i] = (OUT_T)acc;
}

// ------------------------------ WMMA GEMM ----------------------------------
// C[M,N] = A[M,K] @ Bt[N,K]^T   (A bf16 row-major or f32 converted on load,
//                                Bt bf16 N-major, C f32 row-major)
// 256 threads = 8 wave32; waves arranged 4 (M) x 2 (N); wave tile 32x64
// (2x4 tiles of v_wmma_f32_16x16x32_bf16); block tile 128x128.
// B (18.9 MB bf16) is fully resident in the 192 MB L2, so no LDS staging;
// A is the streamed operand and gets an unconditional speculative prefetch
// (failed translations past the end of the buffer are silently dropped).
template <bool A_IS_F32>
__global__ __launch_bounds__(256) void wmma_gemm(const void* __restrict__ Aop,
                                                 const bf16_t* __restrict__ Bt,
                                                 float* __restrict__ C,
                                                 int M, int N, int K) {
  const int lane = threadIdx.x & 31;
  const int wave = threadIdx.x >> 5;
  const int half = lane >> 4;   // 0/1 : lane group per ISA operand layout
  const int l16  = lane & 15;
  const int wm   = wave & 3;    // 0..3
  const int wn   = wave >> 2;   // 0..1
  const int mbase = blockIdx.y * 128 + wm * 32;
  const int nbase = blockIdx.x * 128 + wn * 64;

  v8f acc[2][4];
#pragma unroll
  for (int tm = 0; tm < 2; ++tm)
#pragma unroll
    for (int tn = 0; tn < 4; ++tn)
#pragma unroll
      for (int e = 0; e < 8; ++e) acc[tm][tn][e] = 0.f;

  const float*  Af  = (const float*)Aop;
  const bf16_t* Abf = (const bf16_t*)Aop;

  for (int k0 = 0; k0 < K; k0 += 32) {
    // ---- A fragments: lane holds row m, K = half*8..+7 and 16+half*8..+7
    v16bf afrag[2];
#pragma unroll
    for (int tm = 0; tm < 2; ++tm) {
      const int m = mbase + tm * 16 + l16;
      if (A_IS_F32) {
        const float* pa = Af + (size_t)m * K + k0 + half * 8;
        __builtin_prefetch(pa + 64, 0, 1);   // global_prefetch_b8, 2 K-tiles ahead
        float4 f0 = ((const float4*)pa)[0];
        float4 f1 = ((const float4*)pa)[1];
        float4 f2 = ((const float4*)(pa + 16))[0];
        float4 f3 = ((const float4*)(pa + 16))[1];
        afrag[tm][0]  = (bf16_t)f0.x; afrag[tm][1]  = (bf16_t)f0.y;
        afrag[tm][2]  = (bf16_t)f0.z; afrag[tm][3]  = (bf16_t)f0.w;
        afrag[tm][4]  = (bf16_t)f1.x; afrag[tm][5]  = (bf16_t)f1.y;
        afrag[tm][6]  = (bf16_t)f1.z; afrag[tm][7]  = (bf16_t)f1.w;
        afrag[tm][8]  = (bf16_t)f2.x; afrag[tm][9]  = (bf16_t)f2.y;
        afrag[tm][10] = (bf16_t)f2.z; afrag[tm][11] = (bf16_t)f2.w;
        afrag[tm][12] = (bf16_t)f3.x; afrag[tm][13] = (bf16_t)f3.y;
        afrag[tm][14] = (bf16_t)f3.z; afrag[tm][15] = (bf16_t)f3.w;
      } else {
        const bf16_t* pa = Abf + (size_t)m * K + k0 + half * 8;
        __builtin_prefetch(pa + 64, 0, 1);   // 2 K-tiles ahead
        ((uint4*)&afrag[tm])[0] = ((const uint4*)pa)[0];        // K half*8..+7
        ((uint4*)&afrag[tm])[1] = ((const uint4*)(pa + 16))[0]; // K 16+half*8..+7
      }
    }
    // ---- B fragments: lane holds col n, K = half*8..+7 and 16+half*8..+7
    v16bf bfrag[4];
#pragma unroll
    for (int tn = 0; tn < 4; ++tn) {
      const int n = nbase + tn * 16 + l16;
      const bf16_t* pb = Bt + (size_t)n * K + k0 + half * 8;
      ((uint4*)&bfrag[tn])[0] = ((const uint4*)pb)[0];
      ((uint4*)&bfrag[tn])[1] = ((const uint4*)(pb + 16))[0];
    }
    // ---- 8 WMMAs: D = A x B + C
#pragma unroll
    for (int tm = 0; tm < 2; ++tm)
#pragma unroll
      for (int tn = 0; tn < 4; ++tn)
        acc[tm][tn] = __builtin_amdgcn_wmma_f32_16x16x32_bf16(
            false, afrag[tm], false, bfrag[tn], (short)0, acc[tm][tn],
            false, false);
  }

  // ---- store: C/D layout — VGPR v holds M = half*8 + v, N = l16
#pragma unroll
  for (int tm = 0; tm < 2; ++tm)
#pragma unroll
    for (int tn = 0; tn < 4; ++tn) {
      const int n = nbase + tn * 16 + l16;
#pragma unroll
      for (int v = 0; v < 8; ++v) {
        const int m = mbase + tm * 16 + half * 8 + v;
        C[(size_t)m * N + n] = acc[tm][tn][v];
      }
    }
}

// ------------------------------- launcher ----------------------------------

extern "C" void kernel_launch(void* const* d_in, const int* in_sizes, int n_in,
                              void* d_out, int out_size, void* d_ws, size_t ws_size,
                              hipStream_t stream) {
  const float* x     = (const float*)d_in[0];  // [4,4096,3072]
  const float* U     = (const float*)d_in[1];  // [3072,3072]
  const float* S     = (const float*)d_in[2];  // [3072]
  const float* V     = (const float*)d_in[3];  // [3072,3072]
  const float* delta = (const float*)d_in[4];  // [3072]
  const float* Q1    = (const float*)d_in[5];  // [16,16]
  const float* Q2    = (const float*)d_in[6];  // [16,16]
  const float* Q3    = (const float*)d_in[7];  // [12,12]
  float* out = (float*)d_out;                  // [4,4096,3072]

  const size_t NELEM = (size_t)DIM * DIM;      // 9,437,184
  // Workspace layout (~132 MB total):
  bf16_t* Ubf = (bf16_t*)d_ws;                 // 18.9 MB
  bf16_t* VsT = Ubf + NELEM;                   // 18.9 MB
  float*  M1t = (float*)(VsT + NELEM);         // 37.7 MB
  float*  Tb  = M1t + NELEM;                   // 37.7 MB
  bf16_t* WtT = (bf16_t*)(Tb + NELEM);         // 18.9 MB
  if (ws_size < 5 * NELEM * 4) return;         // deterministic guard

  // 1) U -> bf16
  cvt_f32_bf16<<<(int)((NELEM + 255) / 256), 256, 0, stream>>>(U, Ubf, (int)NELEM);
  // 2) VsT[i,k] = bf16(relu(S+d)[k] * V[k,i])
  scale_transpose<<<dim3(DIM / 32, DIM / 32), dim3(32, 8), 0, stream>>>(V, S, delta, VsT);
  // 3) M1t[o,i] = sum_k Ubf[o,k] * VsT[i,k]   (W^T, transposed to o-major)
  wmma_gemm<false><<<dim3(DIM / 128, DIM / 128), 256, 0, stream>>>(Ubf, VsT, M1t,
                                                                   DIM, DIM, DIM);
  // 4) Kron(Q1,Q2,Q3) applied along i of every row o (3 factored passes)
  kron_stage<float ><<<dim3(DIM / 256, DIM), 256, 0, stream>>>(Q3, M1t, Tb, 12, 1);
  kron_stage<float ><<<dim3(DIM / 256, DIM), 256, 0, stream>>>(Q2, Tb, M1t, 16, 12);
  kron_stage<bf16_t><<<dim3(DIM / 256, DIM), 256, 0, stream>>>(Q1, M1t, WtT, 16, 192);
  // 5) out[r,o] = sum_i x[r,i] * WtT[o,i]
  wmma_gemm<true><<<dim3(DIM / 128, 16384 / 128), 256, 0, stream>>>(x, WtT, out,
                                                                    16384, DIM, DIM);
}
